// SymplecticNet_50457275793840
// MI455X (gfx1250) — compile-verified, matrix operand order checked by reading, not run on hardware
//
#include <hip/hip_runtime.h>
#include <math.h>

// ---------------------------------------------------------------------------
// Types for CDNA5 WMMA
// ---------------------------------------------------------------------------
typedef __attribute__((ext_vector_type(16))) _Float16 v16h;
typedef __attribute__((ext_vector_type(8)))  _Float16 v8h;
typedef __attribute__((ext_vector_type(2)))  _Float16 h2;
typedef __attribute__((ext_vector_type(8)))  float    v8f;

#define D_MODEL 512
#define N_HEADS 8
#define HEAD_DIM 64
#define SEQ_T 512
#define BATCH 32
#define MROWS (BATCH * SEQ_T)   // 16384
#define D_FEAT 158

// ---------------------------------------------------------------------------
// WMMA fragment loader.
// 16-bit A (16x32) lane layout (ISA 7.12.2): lane holds row (lane&15);
// halves 0..7 = K {k0..k0+7}, halves 8..15 = K {k0+16..k0+23}, k0=(lane>=16)?8:0.
// B (32x16) mirrors with lane -> column; B tiles are staged column-major.
// ---------------------------------------------------------------------------
__device__ __forceinline__ v16h frag_h16(const _Float16* tile, int stride,
                                         int rc, int kBase, int lane) {
  int r  = rc + (lane & 15);
  int k0 = kBase + ((lane >> 4) << 3);
  v8h lo = *(const v8h*)(tile + r * stride + k0);
  v8h hi = *(const v8h*)(tile + r * stride + k0 + 16);
  v16h out;
#pragma unroll
  for (int i = 0; i < 8; ++i) { out[i] = lo[i]; out[i + 8] = hi[i]; }
  return out;
}

__device__ __forceinline__ v8f wmma16x16x32(v16h a, v16h b, v8f c) {
  return __builtin_amdgcn_wmma_f32_16x16x32_f16(false, a, false, b,
                                                (short)0, c, false, false);
}

__device__ __forceinline__ float act_apply(float v, int act) {
  if (act == 1) return 0.5f * v * (1.0f + erff(v * 0.70710678118654752f)); // GELU
  if (act == 2) return v / (1.0f + __expf(-v));                            // SiLU
  return v;
}

// ---------------------------------------------------------------------------
// FAST GEMM: C[M,N] = act(A[M,K] @ W[K,N] + bias)
// Requires: M%128==0, N%128==0, K%32==0, lda%4==0, ldw%4==0, W row-major.
// 128x128 block tile, 8 waves of 32x64. K staged 32 wide, f16, software
// pipelined with unguarded b128 loads + prefetch of tile kb+64.
// ---------------------------------------------------------------------------
#define FMT 128
#define FNT 128
#define FKT 32
#define SA_STR 40   // padded strides (halves) -> conflict-free b128 frag reads
#define SB_STR 40

__global__ __launch_bounds__(256)
void k_gemm_fast(const float* __restrict__ A, int lda,
                 const float* __restrict__ W, int ldw,
                 const float* __restrict__ bias,
                 float* __restrict__ C, int ldc, int K, int act) {
  __shared__ __align__(16) _Float16 sA[FMT * SA_STR];
  __shared__ __align__(16) _Float16 sB[FNT * SB_STR];
  const int tid = threadIdx.x, lane = tid & 31, wave = tid >> 5;
  const long bm = (long)blockIdx.x * FMT, bn = (long)blockIdx.y * FNT;
  const int wm = (wave & 3) * 32;
  const int wn = (wave >> 2) * 64;
  // staging coords: A: 128 rows x 32 k (16 k per thread); B: 2 k-rows x 8 n
  const int ar = tid >> 1, ak = (tid & 1) * 16;
  const int bk = (tid >> 4) * 2, bng = (tid & 15) * 8;
  const float* aPtr = A + (bm + ar) * lda + ak;
  const float* bPtr = W + (long)bk * ldw + bn + bng;

  float4 a0, a1, a2, a3, b00, b01, b10, b11;
  auto loadTiles = [&](int kb) {
    const float* ap = aPtr + kb;
    a0 = *(const float4*)(ap + 0);  a1 = *(const float4*)(ap + 4);
    a2 = *(const float4*)(ap + 8);  a3 = *(const float4*)(ap + 12);
    const float* bp = bPtr + (long)kb * ldw;
    b00 = *(const float4*)(bp + 0); b01 = *(const float4*)(bp + 4);
    bp += ldw;
    b10 = *(const float4*)(bp + 0); b11 = *(const float4*)(bp + 4);
  };
  auto storeTiles = [&]() {
    float af[16] = {a0.x, a0.y, a0.z, a0.w, a1.x, a1.y, a1.z, a1.w,
                    a2.x, a2.y, a2.z, a2.w, a3.x, a3.y, a3.z, a3.w};
    v8h h0, h1;
#pragma unroll
    for (int i = 0; i < 8; ++i) { h0[i] = (_Float16)af[i]; h1[i] = (_Float16)af[i + 8]; }
    *(v8h*)(sA + ar * SA_STR + ak)     = h0;
    *(v8h*)(sA + ar * SA_STR + ak + 8) = h1;
    float lo[8] = {b00.x, b00.y, b00.z, b00.w, b01.x, b01.y, b01.z, b01.w};
    float hi[8] = {b10.x, b10.y, b10.z, b10.w, b11.x, b11.y, b11.z, b11.w};
#pragma unroll
    for (int j = 0; j < 8; ++j) {
      h2 t; t[0] = (_Float16)lo[j]; t[1] = (_Float16)hi[j];
      *(h2*)(sB + (bng + j) * SB_STR + bk) = t;   // column-major, k pair packed
    }
  };

  v8f acc[2][4] = {};
  loadTiles(0);
  for (int kb = 0; kb < K; kb += FKT) {
    __syncthreads();
    storeTiles();
    __syncthreads();
    if (kb + FKT < K) loadTiles(kb + FKT);
    if (kb + 2 * FKT < K) {
      __builtin_prefetch(aPtr + kb + 2 * FKT, 0, 1);
      __builtin_prefetch(bPtr + (long)(kb + 2 * FKT) * ldw, 0, 1);
    }
    v16h afr[2], bfr[4];
#pragma unroll
    for (int mi = 0; mi < 2; ++mi) afr[mi] = frag_h16(sA, SA_STR, wm + mi * 16, 0, lane);
#pragma unroll
    for (int ni = 0; ni < 4; ++ni) bfr[ni] = frag_h16(sB, SB_STR, wn + ni * 16, 0, lane);
#pragma unroll
    for (int mi = 0; mi < 2; ++mi)
#pragma unroll
      for (int ni = 0; ni < 4; ++ni)
        acc[mi][ni] = wmma16x16x32(afr[mi], bfr[ni], acc[mi][ni]);
  }

  const int mo = (lane >> 4) * 8, nc = lane & 15;
#pragma unroll
  for (int mi = 0; mi < 2; ++mi)
#pragma unroll
    for (int ni = 0; ni < 4; ++ni) {
      long col = bn + wn + ni * 16 + nc;
      float bval = bias ? bias[col] : 0.0f;
#pragma unroll
      for (int r = 0; r < 8; ++r) {
        long row = bm + wm + mi * 16 + r + mo;
        C[row * ldc + col] = act_apply(acc[mi][ni][r] + bval, act);
      }
    }
}

// ---------------------------------------------------------------------------
// Generic GEMM (small / ragged shapes): branchless clamped-index staging so
// loads stay unconditional and batchable. 64x128 tile, 8 waves of 32x32.
// ---------------------------------------------------------------------------
#define GMT 64
#define GNT 128
#define GKT 32

__global__ __launch_bounds__(256)
void k_gemm(const float* __restrict__ A, int lda,
            const float* __restrict__ W, int ldw, int wTrans,
            const float* __restrict__ bias,
            float* __restrict__ C, int ldc,
            int M, int N, int K, int act, int addC) {
  __shared__ __align__(16) _Float16 sA[GMT * SA_STR];
  __shared__ __align__(16) _Float16 sB[GNT * SB_STR];
  const int tid = threadIdx.x, lane = tid & 31, wave = tid >> 5;
  const int bm = blockIdx.x * GMT, bn = blockIdx.y * GNT;
  const int wm = (wave & 1) * 32;
  const int wn = (wave >> 1) * 32;

  v8f acc[2][2] = {};
  for (int kb = 0; kb < K; kb += GKT) {
    __syncthreads();
    {   // A: row tid>>2, 8 ks
      int r = tid >> 2, kk = (tid & 3) * 8;
      int gr = bm + r;
#pragma unroll
      for (int j = 0; j < 8; ++j) {
        int gk = kb + kk + j;
        bool ok = (gr < M) && (gk < K);
        long idx = ok ? ((long)gr * lda + gk) : 0;
        float v = A[idx];
        sA[r * SA_STR + kk + j] = (_Float16)(ok ? v : 0.0f);
      }
    }
    {   // B: col tid>>1, 16 ks, column-major stage
      int n = tid >> 1, kk = (tid & 1) * 16;
      int gn = bn + n;
#pragma unroll
      for (int j = 0; j < 16; ++j) {
        int gk = kb + kk + j;
        bool ok = (gn < N) && (gk < K);
        long idx = ok ? (wTrans ? ((long)gn * ldw + gk) : ((long)gk * ldw + gn)) : 0;
        float v = W[idx];
        sB[n * SB_STR + kk + j] = (_Float16)(ok ? v : 0.0f);
      }
    }
    __syncthreads();
    v16h af0 = frag_h16(sA, SA_STR, wm +  0, 0, lane);
    v16h af1 = frag_h16(sA, SA_STR, wm + 16, 0, lane);
    v16h bf0 = frag_h16(sB, SB_STR, wn +  0, 0, lane);
    v16h bf1 = frag_h16(sB, SB_STR, wn + 16, 0, lane);
    acc[0][0] = wmma16x16x32(af0, bf0, acc[0][0]);
    acc[0][1] = wmma16x16x32(af0, bf1, acc[0][1]);
    acc[1][0] = wmma16x16x32(af1, bf0, acc[1][0]);
    acc[1][1] = wmma16x16x32(af1, bf1, acc[1][1]);
  }

  const int mo = (lane >> 4) * 8, nc = lane & 15;
#pragma unroll
  for (int mi = 0; mi < 2; ++mi)
#pragma unroll
    for (int ni = 0; ni < 2; ++ni) {
      int col = bn + wn + ni * 16 + nc;
      if (col >= N) continue;
      float bval = bias ? bias[col] : 0.0f;
#pragma unroll
      for (int r = 0; r < 8; ++r) {
        int row = bm + wm + mi * 16 + r + mo;
        if (row >= M) continue;
        float v = act_apply(acc[mi][ni][r] + bval, act);
        long idx = (long)row * ldc + col;
        if (addC) v += C[idx];
        C[idx] = v;
      }
    }
}

// ---------------------------------------------------------------------------
// Attention: one block per (b, h, 64-row q tile). Full 64x512 f32 score row in
// LDS (CDNA5 320KB budget). scores = (q*0.125)@k^T via WMMA, block softmax
// writes f16 weights to a padded LDS buffer, attn = wgt@v via WMMA.
// ---------------------------------------------------------------------------
#define SQ_STR 72          // halves; conflict-free
#define SV_STR 40
#define SW_STR 520
#define ATT_SMEM (64*SQ_STR*2 /*sQ*/ + 64*SQ_STR*2 /*sB*/ + 64*512*4 /*sS*/ \
                  + 64*SW_STR*2 /*sW*/)   // = 216064 bytes

__global__ __launch_bounds__(256)
void k_attn(const float* __restrict__ qkv, float* __restrict__ out) {
  extern __shared__ __align__(16) char smem[];
  _Float16* sQ = (_Float16*)smem;                        // [64][72]
  _Float16* sB = sQ + 64 * SQ_STR;                       // [64][72] (K:72 / V:40)
  float*    sS = (float*)(smem + 64 * SQ_STR * 2 * 2);   // [64][512]
  _Float16* sW = (_Float16*)(smem + 64 * SQ_STR * 2 * 2 + 64 * 512 * 4); // [64][520]

  const int tid = threadIdx.x, lane = tid & 31, wave = tid >> 5;
  const int blk = blockIdx.x;
  const int qt = blk & 7, h = (blk >> 3) & 7, b = blk >> 6;
  const long ldq = 3 * D_MODEL;
  const int rowBase = b * SEQ_T + qt * 64;

  // ---- load Q tile, scale 1/sqrt(64) folded ----
  {
    int r = tid >> 2, d0 = (tid & 3) * 16;
    const float* qp = qkv + (long)(rowBase + r) * ldq + h * HEAD_DIM + d0;
    float4 q0 = *(const float4*)(qp + 0),  q1 = *(const float4*)(qp + 4);
    float4 q2 = *(const float4*)(qp + 8),  q3 = *(const float4*)(qp + 12);
    float qf[16] = {q0.x, q0.y, q0.z, q0.w, q1.x, q1.y, q1.z, q1.w,
                    q2.x, q2.y, q2.z, q2.w, q3.x, q3.y, q3.z, q3.w};
    v8h h0, h1;
#pragma unroll
    for (int i = 0; i < 8; ++i) {
      h0[i] = (_Float16)(qf[i] * 0.125f);
      h1[i] = (_Float16)(qf[i + 8] * 0.125f);
    }
    *(v8h*)(sQ + r * SQ_STR + d0)     = h0;
    *(v8h*)(sQ + r * SQ_STR + d0 + 8) = h1;
  }

  // ---- scores = Q @ K^T, per 64-key tile ----
  {
    const int wm = (wave & 3) * 16, wn = (wave >> 2) * 32;
    for (int kt = 0; kt < 8; ++kt) {
      __syncthreads();
      {
        int r = tid >> 2, d0 = (tid & 3) * 16;
        const float* kp = qkv + (long)(b * SEQ_T + kt * 64 + r) * ldq
                          + D_MODEL + h * HEAD_DIM + d0;
        float4 k0 = *(const float4*)(kp + 0), k1 = *(const float4*)(kp + 4);
        float4 k2 = *(const float4*)(kp + 8), k3 = *(const float4*)(kp + 12);
        float kf[16] = {k0.x, k0.y, k0.z, k0.w, k1.x, k1.y, k1.z, k1.w,
                        k2.x, k2.y, k2.z, k2.w, k3.x, k3.y, k3.z, k3.w};
        v8h h0, h1;
#pragma unroll
        for (int i = 0; i < 8; ++i) { h0[i] = (_Float16)kf[i]; h1[i] = (_Float16)kf[i + 8]; }
        *(v8h*)(sB + r * SQ_STR + d0)     = h0;
        *(v8h*)(sB + r * SQ_STR + d0 + 8) = h1;
      }
      __syncthreads();
      v8f acc[2] = {};
#pragma unroll
      for (int kb = 0; kb < 64; kb += 32) {
        v16h af  = frag_h16(sQ, SQ_STR, wm, kb, lane);
        v16h bf0 = frag_h16(sB, SQ_STR, wn +  0, kb, lane);
        v16h bf1 = frag_h16(sB, SQ_STR, wn + 16, kb, lane);
        acc[0] = wmma16x16x32(af, bf0, acc[0]);
        acc[1] = wmma16x16x32(af, bf1, acc[1]);
      }
      const int mo = (lane >> 4) * 8, nc = lane & 15;
#pragma unroll
      for (int ni = 0; ni < 2; ++ni)
#pragma unroll
        for (int r = 0; r < 8; ++r)
          sS[(wm + r + mo) * SEQ_T + kt * 64 + wn + ni * 16 + nc] = acc[ni][r];
    }
  }
  __syncthreads();

  // ---- softmax over 512 keys (4 lanes/row); write f16 weights to sW ----
  {
    int row = tid >> 2, seg = (tid & 3) * 128;
    float* rp = sS + row * SEQ_T + seg;
    float mx = -3.4e38f;
    for (int j = 0; j < 128; ++j) mx = fmaxf(mx, rp[j]);
    mx = fmaxf(mx, __shfl_xor(mx, 1, 32));
    mx = fmaxf(mx, __shfl_xor(mx, 2, 32));
    float sum = 0.0f;
    for (int j = 0; j < 128; ++j) { float e = __expf(rp[j] - mx); rp[j] = e; sum += e; }
    sum += __shfl_xor(sum, 1, 32);
    sum += __shfl_xor(sum, 2, 32);
    float inv = 1.0f / sum;
    _Float16* wp = sW + row * SW_STR + seg;
    for (int j = 0; j < 128; j += 2) {
      h2 t; t[0] = (_Float16)(rp[j] * inv); t[1] = (_Float16)(rp[j + 1] * inv);
      *(h2*)(wp + j) = t;
    }
  }

  // ---- attn = wgt @ V ----
  {
    const int wm = (wave & 3) * 16, wn = (wave >> 2) * 32;
    v8f acc[2] = {};
    for (int kt = 0; kt < 16; ++kt) {   // 32 keys per step
      __syncthreads();
      {   // stage V rows kt*32..+31 transposed into sB[d][kj], k-pairs packed
        int kj = (tid >> 4) * 2, dg = (tid & 15) * 4;
        const float* vp = qkv + (long)(b * SEQ_T + kt * 32 + kj) * ldq
                          + 2 * D_MODEL + h * HEAD_DIM + dg;
        float4 v0 = *(const float4*)(vp);
        float4 v1 = *(const float4*)(vp + ldq);
        float lo[4] = {v0.x, v0.y, v0.z, v0.w};
        float hi[4] = {v1.x, v1.y, v1.z, v1.w};
#pragma unroll
        for (int j = 0; j < 4; ++j) {
          h2 t; t[0] = (_Float16)lo[j]; t[1] = (_Float16)hi[j];
          *(h2*)(sB + (dg + j) * SV_STR + kj) = t;
        }
      }
      __syncthreads();
      v16h af  = frag_h16(sW, SW_STR, wm, kt * 32, lane);
      v16h bf0 = frag_h16(sB, SV_STR, wn +  0, 0, lane);
      v16h bf1 = frag_h16(sB, SV_STR, wn + 16, 0, lane);
      acc[0] = wmma16x16x32(af, bf0, acc[0]);
      acc[1] = wmma16x16x32(af, bf1, acc[1]);
    }
    const int mo = (lane >> 4) * 8, nc = lane & 15;
#pragma unroll
    for (int ni = 0; ni < 2; ++ni)
#pragma unroll
      for (int r = 0; r < 8; ++r) {
        int row = rowBase + wm + r + mo;
        int col = h * HEAD_DIM + wn + ni * 16 + nc;
        out[(long)row * D_MODEL + col] = acc[ni][r];
      }
  }
}

// ---------------------------------------------------------------------------
// Elementwise / reduction kernels (f32 glue)
// ---------------------------------------------------------------------------
__global__ void k_fracdiff(const float* __restrict__ x, float* __restrict__ y) {
  int idx = blockIdx.x * 256 + threadIdx.x;
  if (idx >= BATCH * SEQ_T * D_FEAT) return;
  int f = idx % D_FEAT;
  int t = (idx / D_FEAT) % SEQ_T;
  int b = idx / (D_FEAT * SEQ_T);
  float acc = 0.0f, c = 1.0f;
  for (int k = 0; k < 64; ++k) {
    if (k > 0) c *= (0.12f - (float)k + 1.0f) / (float)k;
    int tt = t - k;
    if (tt >= 0) acc += c * x[((long)b * SEQ_T + tt) * D_FEAT + f];
  }
  y[idx] = acc;
}

__global__ __launch_bounds__(256)
void k_ln(const float* __restrict__ X, const float* __restrict__ g,
          const float* __restrict__ bta, float* __restrict__ Y, int rows) {
  int row = blockIdx.x * 8 + (threadIdx.x >> 5);
  int lane = threadIdx.x & 31;
  if (row >= rows) return;
  const float* x = X + (long)row * D_MODEL;
  float s = 0.0f, ss = 0.0f;
  for (int j = lane; j < D_MODEL; j += 32) { float v = x[j]; s += v; ss += v * v; }
#pragma unroll
  for (int o = 16; o > 0; o >>= 1) { s += __shfl_xor(s, o, 32); ss += __shfl_xor(ss, o, 32); }
  float m = s * (1.0f / D_MODEL);
  float var = ss * (1.0f / D_MODEL) - m * m;
  float inv = rsqrtf(var + 1e-5f);
  float* yo = Y + (long)row * D_MODEL;
  for (int j = lane; j < D_MODEL; j += 32) yo[j] = (x[j] - m) * inv * g[j] + bta[j];
}

__global__ void k_sym(const float* __restrict__ a, float* __restrict__ s, int n) {
  int idx = blockIdx.x * 256 + threadIdx.x;
  if (idx >= n) return;
  int d = idx & (D_MODEL - 1);
  long row = idx >> 9;
  s[idx] = (d < 256) ? a[row * D_MODEL + 256 + d] : -a[row * D_MODEL + d - 256];
}

__global__ void k_colmean(const float* __restrict__ cons, float* __restrict__ cm) {
  int idx = blockIdx.x * 256 + threadIdx.x;
  if (idx >= BATCH * D_MODEL) return;
  int d = idx & (D_MODEL - 1);
  int b = idx >> 9;
  float s = 0.0f;
  for (int t = 0; t < SEQ_T; ++t) s += cons[((long)b * SEQ_T + t) * D_MODEL + d];
  cm[idx] = s * (1.0f / SEQ_T);
}

__global__ void k_resid_sub(float* __restrict__ h, const float* __restrict__ out,
                            const float* __restrict__ cm, int n) {
  int idx = blockIdx.x * 256 + threadIdx.x;
  if (idx >= n) return;
  int d = idx & (D_MODEL - 1);
  int b = idx >> 18;
  h[idx] += out[idx] - cm[b * D_MODEL + d];
}

__global__ void k_add(float* __restrict__ h, const float* __restrict__ f, int n) {
  int idx = blockIdx.x * 256 + threadIdx.x;
  if (idx < n) h[idx] += f[idx];
}

__global__ void k_meanT(const float* __restrict__ h, float* __restrict__ z) {
  int idx = blockIdx.x * 256 + threadIdx.x;
  if (idx >= BATCH * D_MODEL) return;
  int d = idx & (D_MODEL - 1);
  int b = idx >> 9;
  float s = 0.0f;
  for (int t = 0; t < SEQ_T; ++t) s += h[((long)b * SEQ_T + t) * D_MODEL + d];
  z[idx] = s * (1.0f / SEQ_T);
}

__global__ void k_zero(float* p, int n) {
  int idx = blockIdx.x * 256 + threadIdx.x;
  if (idx < n) p[idx] = 0.0f;
}

__global__ void k_state(const float* __restrict__ z, const float* __restrict__ tb,
                        float* __restrict__ st) {
  int idx = blockIdx.x * 256 + threadIdx.x;
  if (idx >= BATCH * 544) return;
  int j = idx % 544;
  int b = idx / 544;
  st[idx] = (j < 512) ? z[b * D_MODEL + j] : ((j == 512) ? tb[b] : 0.0f);
}

__global__ void k_hamupd(float* __restrict__ z, const float* __restrict__ hv,
                         float* __restrict__ tb) {
  int idx = blockIdx.x * 256 + threadIdx.x;
  if (idx >= BATCH * 256) return;
  int d = idx & 255;
  int b = idx >> 8;
  float gate = 1.0f / (1.0f + __expf(-hv[b]));
  float q = z[b * D_MODEL + d];
  float p = z[b * D_MODEL + 256 + d];
  float pn = p - 0.1f * tanhf(q) * gate;
  float qn = q + 0.1f * tanhf(p) * gate;
  z[b * D_MODEL + d] = qn;
  z[b * D_MODEL + 256 + d] = pn;
  if (d == 0) tb[b] += 0.1f;
}

__global__ __launch_bounds__(256)
void k_softmax(float* __restrict__ E, int rows, float temp) {
  int row = blockIdx.x * 8 + (threadIdx.x >> 5);
  int lane = threadIdx.x & 31;
  if (row >= rows) return;
  float* e = E + (long)row * 512;
  float mx = -3.4e38f;
  for (int j = lane; j < 512; j += 32) mx = fmaxf(mx, e[j] * temp);
#pragma unroll
  for (int o = 16; o > 0; o >>= 1) mx = fmaxf(mx, __shfl_xor(mx, o, 32));
  float s = 0.0f;
  for (int j = lane; j < 512; j += 32) { float v = __expf(e[j] * temp - mx); e[j] = v; s += v; }
#pragma unroll
  for (int o = 16; o > 0; o >>= 1) s += __shfl_xor(s, o, 32);
  float inv = 1.0f / s;
  for (int j = lane; j < 512; j += 32) e[j] *= inv;
}

// ---------------------------------------------------------------------------
// Host orchestration
// ---------------------------------------------------------------------------
static inline void gemm(hipStream_t s, const float* A, int lda, const float* W,
                        int ldw, int wT, const float* bias, float* C, int ldc,
                        int M, int N, int K, int act, int addC) {
  bool fast = !wT && !addC && (M % FMT == 0) && (N % FNT == 0) && (K % FKT == 0)
              && (lda % 4 == 0) && (ldw % 4 == 0);
  if (fast) {
    dim3 grid(M / FMT, N / FNT);
    k_gemm_fast<<<grid, 256, 0, s>>>(A, lda, W, ldw, bias, C, ldc, K, act);
  } else {
    dim3 grid((M + GMT - 1) / GMT, (N + GNT - 1) / GNT);
    k_gemm<<<grid, 256, 0, s>>>(A, lda, W, ldw, wT, bias, C, ldc, M, N, K, act, addC);
  }
}

extern "C" void kernel_launch(void* const* d_in, const int* in_sizes, int n_in,
                              void* d_out, int out_size, void* d_ws, size_t ws_size,
                              hipStream_t stream) {
  (void)in_sizes; (void)n_in; (void)out_size; (void)ws_size;
  #define P(i) ((const float*)d_in[i])
  const float* x = P(0);
  float* ws = (float*)d_ws;
  size_t off = 0;
  auto alloc = [&](size_t n) { float* p = ws + off; off += n; return p; };
  float* xfd    = alloc((size_t)MROWS * D_FEAT);
  float* hbuf   = alloc((size_t)MROWS * D_MODEL);
  float* t0     = alloc((size_t)MROWS * 2048);
  float* t1     = alloc((size_t)MROWS * D_MODEL);
  float* t2     = alloc((size_t)MROWS * D_MODEL);
  float* t3     = alloc((size_t)MROWS * D_MODEL);
  float* cmean  = alloc((size_t)BATCH * D_MODEL);
  float* zbuf   = alloc((size_t)BATCH * D_MODEL);
  float* state  = alloc((size_t)BATCH * 544);
  float* hh1    = alloc((size_t)BATCH * 1024);
  float* hh2    = alloc((size_t)BATCH * 1024);
  float* hv     = alloc((size_t)BATCH);
  float* tb     = alloc((size_t)BATCH);
  float* energy = alloc((size_t)BATCH * 512);
  float* y1     = alloc((size_t)BATCH * D_MODEL);
  float* y2     = alloc((size_t)BATCH * 256);

  const int NE = MROWS * D_MODEL;

  // 1. fractional differencing
  k_fracdiff<<<(MROWS * D_FEAT + 255) / 256, 256, 0, stream>>>(x, xfd);
  // 2. input projection + LN
  gemm(stream, xfd, D_FEAT, P(1), D_MODEL, 0, P(2), hbuf, D_MODEL,
       MROWS, D_MODEL, D_FEAT, 0, 0);
  k_ln<<<MROWS / 8, 256, 0, stream>>>(hbuf, P(3), P(4), hbuf, MROWS);

  // 3. transformer layers
  for (int l = 0; l < 4; ++l) {
    int pb = 5 + 16 * l;
    k_ln<<<MROWS / 8, 256, 0, stream>>>(hbuf, P(pb + 0), P(pb + 1), t1, MROWS);
    gemm(stream, t1, D_MODEL, P(pb + 4), 3 * D_MODEL, 0, P(pb + 5), t0,
         3 * D_MODEL, MROWS, 3 * D_MODEL, D_MODEL, 0, 0);
    k_attn<<<BATCH * N_HEADS * (SEQ_T / 64), 256, ATT_SMEM, stream>>>(t0, t2);
    k_sym<<<(NE + 255) / 256, 256, 0, stream>>>(t2, t3, NE);
    gemm(stream, t3, D_MODEL, P(pb + 6), D_MODEL, 0, P(pb + 7), t1, D_MODEL,
         MROWS, D_MODEL, D_MODEL, 0, 0);
    gemm(stream, t1, D_MODEL, P(pb + 8), D_MODEL, 0, P(pb + 9), t2, D_MODEL,
         MROWS, D_MODEL, D_MODEL, 1, 0);
    gemm(stream, t2, D_MODEL, P(pb + 10), D_MODEL, 0, P(pb + 11), t3, D_MODEL,
         MROWS, D_MODEL, D_MODEL, 0, 0);
    k_colmean<<<(BATCH * D_MODEL + 255) / 256, 256, 0, stream>>>(t3, cmean);
    k_resid_sub<<<(NE + 255) / 256, 256, 0, stream>>>(hbuf, t1, cmean, NE);
    k_ln<<<MROWS / 8, 256, 0, stream>>>(hbuf, P(pb + 2), P(pb + 3), t1, MROWS);
    gemm(stream, t1, D_MODEL, P(pb + 12), 2048, 0, P(pb + 13), t0, 2048,
         MROWS, 2048, D_MODEL, 1, 0);
    gemm(stream, t0, 2048, P(pb + 14), D_MODEL, 0, P(pb + 15), t1, D_MODEL,
         MROWS, D_MODEL, 2048, 0, 0);
    k_add<<<(NE + 255) / 256, 256, 0, stream>>>(hbuf, t1, NE);
  }

  // 4. z = mean over T
  k_meanT<<<(BATCH * D_MODEL + 255) / 256, 256, 0, stream>>>(hbuf, zbuf);

  // 5. Hamiltonian ODE
  k_zero<<<1, 256, 0, stream>>>(tb, BATCH);
  for (int s5 = 0; s5 < 5; ++s5) {
    k_state<<<(BATCH * 544 + 255) / 256, 256, 0, stream>>>(zbuf, tb, state);
    gemm(stream, state, 544, P(69), 1024, 0, P(70), hh1, 1024,
         BATCH, 1024, 513, 2, 0);
    gemm(stream, hh1, 1024, P(71), 1024, 0, P(72), hh2, 1024,
         BATCH, 1024, 1024, 2, 0);
    gemm(stream, hh2, 1024, P(73), 1, 0, P(74), hv, 1,
         BATCH, 1, 1024, 0, 0);
    k_hamupd<<<(BATCH * 256 + 255) / 256, 256, 0, stream>>>(zbuf, hv, tb);
  }

  // 6. holographic memory (residual into zbuf)
  gemm(stream, zbuf, D_MODEL, P(75), 256, 1, nullptr, energy, 512,
       BATCH, 512, 256, 0, 0);
  gemm(stream, zbuf + 256, D_MODEL, P(76), 256, 1, nullptr, energy, 512,
       BATCH, 512, 256, 0, 1);
  k_softmax<<<BATCH / 8, 256, 0, stream>>>(energy, BATCH, 0.0625f);
  gemm(stream, energy, 512, P(77), 256, 0, nullptr, zbuf, D_MODEL,
       BATCH, 256, 512, 0, 1);
  gemm(stream, energy, 512, P(78), 256, 0, nullptr, zbuf + 256, D_MODEL,
       BATCH, 256, 512, 0, 1);

  // 7. head
  k_ln<<<BATCH / 8, 256, 0, stream>>>(zbuf, P(79), P(80), y1, BATCH);
  gemm(stream, y1, D_MODEL, P(81), 256, 0, P(82), y2, 256,
       BATCH, 256, D_MODEL, 1, 0);
  gemm(stream, y2, 256, P(83), 1, 0, P(84), (float*)d_out, 1,
       BATCH, 1, 256, 0, 0);
  #undef P
}
